// Net_17600775979412
// MI455X (gfx1250) — matrix operand order repaired and checked by reading, not compiled, not run on hardware
//
#include <hip/hip_runtime.h>
#include <math.h>

// ---------------- problem constants (from reference) ----------------
#define NN    100000
#define EE    1600000
#define FIN   512
#define HID   256
#define NCLS  47
#define NPAD  64      // padded class dim for GEMM2 / aggregation buffers

// ---------------- WMMA vector types (gfx1250, wave32) ----------------
typedef __attribute__((ext_vector_type(16))) __bf16 v16bf;
typedef __attribute__((ext_vector_type(8)))  __bf16 v8bf;
typedef __attribute__((ext_vector_type(8)))  float  v8f;

// =====================================================================
// gcn_norm pieces
// =====================================================================
__global__ void k_deg_init(float* deg) {
    int i = blockIdx.x * blockDim.x + threadIdx.x;
    if (i < NN) deg[i] = 1.0f;   // self-loop weight
}

__global__ void k_deg_scatter(const int* __restrict__ ei,
                              const float* __restrict__ ew,
                              float* __restrict__ deg) {
    int e = blockIdx.x * blockDim.x + threadIdx.x;
    if (e < EE) atomicAdd(&deg[ei[EE + e]], ew[e]);   // col = ei[1][e]
}

__global__ void k_dinv(float* deg_dinv) {
    int i = blockIdx.x * blockDim.x + threadIdx.x;
    if (i < NN) {
        float d = deg_dinv[i];
        deg_dinv[i] = (d > 0.0f) ? rsqrtf(d) : 0.0f;
    }
}

__global__ void k_norm(const int* __restrict__ ei,
                       const float* __restrict__ ew,
                       const float* __restrict__ dinv,
                       float* __restrict__ norm) {
    int e = blockIdx.x * blockDim.x + threadIdx.x;
    if (e < EE) norm[e] = dinv[ei[e]] * ew[e] * dinv[ei[EE + e]];
}

// =====================================================================
// weight transpose + f32 -> bf16  (W [K][Nsrc] row-major  ->  Wt [Npad][K])
// =====================================================================
__global__ void k_convW(const float* __restrict__ W, __bf16* __restrict__ Wt,
                        int K, int Nsrc, int Npadded) {
    int idx = blockIdx.x * blockDim.x + threadIdx.x;
    if (idx >= Npadded * K) return;
    int n = idx / K, k = idx - n * K;
    float v = (n < Nsrc) ? W[(size_t)k * Nsrc + n] : 0.0f;
    Wt[(size_t)n * K + k] = (__bf16)v;
}

// =====================================================================
// WMMA GEMM:  C[M][NC] = A[M][K] (f32, cvt->bf16) @ Bt[NC][K] (bf16)
// block = 256 thr = 8 waves (4 M x 2 N), tile 128 x 64, BK = 32
// OOB M rows are *clamped* (not zeroed): they only feed C rows that are
// never stored, so the hot loop has no divergent branch.
// =====================================================================
#define LDS_STRIDE 48   // bf16 elems; 96B rows keep 16B alignment, low conflict

__global__ __launch_bounds__(256)
void k_gemm_bf16(const float* __restrict__ A, const __bf16* __restrict__ Bt,
                 float* __restrict__ C, int M, int K, int NC) {
    __shared__ __bf16 As[128][LDS_STRIDE];
    __shared__ __bf16 Bs[64][LDS_STRIDE];

    const int t      = threadIdx.x;
    const int lane   = t & 31;
    const int wid    = t >> 5;
    const int wave_m = wid >> 1;          // 0..3
    const int wave_n = wid & 1;           // 0..1
    const int g      = lane >> 4;         // lane half
    const int l16    = lane & 15;

    const int blockM = blockIdx.x * 128;
    const int blockN = blockIdx.y * 64;

    v8f acc[2][2];
    #pragma unroll
    for (int sm = 0; sm < 2; ++sm)
        #pragma unroll
        for (int sn = 0; sn < 2; ++sn)
            #pragma unroll
            for (int j = 0; j < 8; ++j) acc[sm][sn][j] = 0.0f;

    // staging coordinates
    const int arow = t >> 1,  ac0 = (t & 1) * 16;   // A: 128 rows x 32, 16 f32/thread
    const int brow = t >> 2,  bc0 = (t & 3) * 8;    // B: 64 rows x 32, 8 bf16/thread

    // clamp OOB rows instead of branching (their outputs are never stored)
    int gm = blockM + arow;
    if (gm >= M) gm = M - 1;
    const float*  asrc = A  + (size_t)gm * K + ac0;
    const __bf16* bsrc = Bt + (size_t)(blockN + brow) * K + bc0;

    for (int k0 = 0; k0 < K; k0 += 32) {
        // ---- stage A (f32 -> bf16), branch-free ----
        {
            const float* src = asrc + k0;
            #pragma unroll
            for (int j = 0; j < 16; ++j) As[arow][ac0 + j] = (__bf16)src[j];
            if (k0 + 32 < K) __builtin_prefetch(src + 32, 0, 1);
        }
        // ---- stage B (pre-converted bf16, [n][K]) ----
        {
            const __bf16* src = bsrc + k0;
            #pragma unroll
            for (int j = 0; j < 8; ++j) Bs[brow][bc0 + j] = src[j];
        }
        __syncthreads();

        // ---- fragments per ISA VGPR layouts ----
        v16bf afr[2], bfr[2];
        #pragma unroll
        for (int sm = 0; sm < 2; ++sm) {
            int r = wave_m * 32 + sm * 16 + l16;       // M row in tile
            v8bf lo = *(const v8bf*)&As[r][g * 8];
            v8bf hi = *(const v8bf*)&As[r][16 + g * 8];
            #pragma unroll
            for (int j = 0; j < 8; ++j) { afr[sm][j] = lo[j]; afr[sm][8 + j] = hi[j]; }
        }
        #pragma unroll
        for (int sn = 0; sn < 2; ++sn) {
            int r = wave_n * 32 + sn * 16 + l16;       // N row (Bt transposed)
            v8bf lo = *(const v8bf*)&Bs[r][g * 8];
            v8bf hi = *(const v8bf*)&Bs[r][16 + g * 8];
            #pragma unroll
            for (int j = 0; j < 8; ++j) { bfr[sn][j] = lo[j]; bfr[sn][8 + j] = hi[j]; }
        }

        #pragma unroll
        for (int sm = 0; sm < 2; ++sm)
            #pragma unroll
            for (int sn = 0; sn < 2; ++sn)
                acc[sm][sn] = __builtin_amdgcn_wmma_f32_16x16x32_bf16(
                    false, afr[sm], false, bfr[sn], (short)0, acc[sm][sn],
                    false, false);

        __syncthreads();
    }

    // ---- store C (f32 16x16 layout: M = r + g*8, N = l16) ----
    #pragma unroll
    for (int sm = 0; sm < 2; ++sm) {
        #pragma unroll
        for (int r = 0; r < 8; ++r) {
            int gmo = blockM + wave_m * 32 + sm * 16 + g * 8 + r;
            if (gmo < M) {
                #pragma unroll
                for (int sn = 0; sn < 2; ++sn)
                    C[(size_t)gmo * NC + blockN + wave_n * 32 + sn * 16 + l16] =
                        acc[sm][sn][r];
            }
        }
    }
}

// =====================================================================
// aggregation:  h = t * dinv^2 (self-loop)  -> edge scatter  -> epilogue
// =====================================================================
__global__ void k_selfloop_init(const float* __restrict__ t,
                                const float* __restrict__ dinv,
                                float* __restrict__ h, int C) {
    size_t idx = (size_t)blockIdx.x * blockDim.x + threadIdx.x;
    if (idx >= (size_t)NN * C) return;
    size_t i = idx / C;
    float s = dinv[i] * dinv[i];
    h[idx] = t[idx] * s;
}

// one block per edge; blockDim = C (256) or NPAD (64)
__global__ void k_scatter(const float* __restrict__ t, float* __restrict__ h,
                          const int* __restrict__ ei,
                          const float* __restrict__ norm,
                          int C, int Cvalid) {
    int e = blockIdx.x;
    int f = threadIdx.x;
    if (f >= Cvalid) return;
    int r = ei[e];
    int c = ei[EE + e];
    float nm = norm[e];
    atomicAdd(&h[(size_t)c * C + f], t[(size_t)r * C + f] * nm);
}

__global__ void k_bias_relu(float* __restrict__ h, const float* __restrict__ b) {
    size_t idx = (size_t)blockIdx.x * blockDim.x + threadIdx.x;
    if (idx >= (size_t)NN * HID) return;
    int f = (int)(idx & (HID - 1));
    h[idx] = fmaxf(h[idx] + b[f], 0.0f);
}

__global__ void k_logsoftmax(const float* __restrict__ agg,
                             const float* __restrict__ b,
                             float* __restrict__ out) {
    int i = blockIdx.x * blockDim.x + threadIdx.x;
    if (i >= NN) return;
    float v[NCLS];
    float m = -3.402823e38f;
    #pragma unroll
    for (int c = 0; c < NCLS; ++c) {
        v[c] = agg[(size_t)i * NPAD + c] + b[c];
        m = fmaxf(m, v[c]);
    }
    float s = 0.0f;
    #pragma unroll
    for (int c = 0; c < NCLS; ++c) s += __expf(v[c] - m);
    float l = m + logf(s);
    #pragma unroll
    for (int c = 0; c < NCLS; ++c) out[(size_t)i * NCLS + c] = v[c] - l;
}

// =====================================================================
// launcher
// =====================================================================
extern "C" void kernel_launch(void* const* d_in, const int* in_sizes, int n_in,
                              void* d_out, int out_size, void* d_ws, size_t ws_size,
                              hipStream_t stream) {
    const float* x   = (const float*)d_in[0];
    const int*   ei  = (const int*)d_in[1];     // (2, E): row = ei[e], col = ei[E+e]
    const float* ew  = (const float*)d_in[2];
    const float* W1  = (const float*)d_in[3];
    const float* b1  = (const float*)d_in[4];
    const float* W2  = (const float*)d_in[5];
    const float* b2  = (const float*)d_in[6];
    float* out = (float*)d_out;

    // -------- workspace carve-up --------
    char* p = (char*)d_ws;
    size_t off = 0;
    auto take = [&](size_t bytes) -> char* {
        char* cur = p + off;
        off = (off + bytes + 255) & ~(size_t)255;
        return cur;
    };
    float*  dinv = (float*)take((size_t)NN * 4);           // deg -> dinv in place
    float*  nrm  = (float*)take((size_t)EE * 4);
    __bf16* W1t  = (__bf16*)take((size_t)HID * FIN * 2);   // [256][512]
    __bf16* W2t  = (__bf16*)take((size_t)NPAD * HID * 2);  // [64][256]
    float*  t1   = (float*)take((size_t)NN * HID * 4);     // x @ W1
    float*  h1   = (float*)take((size_t)NN * HID * 4);     // aggregated + relu
    float*  t2   = (float*)take((size_t)NN * NPAD * 4);    // h1 @ W2
    float*  ag2  = (float*)take((size_t)NN * NPAD * 4);    // aggregated layer2

    const int TPB = 256;
    // ---- norm ----
    k_deg_init   <<<(NN + TPB - 1) / TPB, TPB, 0, stream>>>(dinv);
    k_deg_scatter<<<(EE + TPB - 1) / TPB, TPB, 0, stream>>>(ei, ew, dinv);
    k_dinv       <<<(NN + TPB - 1) / TPB, TPB, 0, stream>>>(dinv);
    k_norm       <<<(EE + TPB - 1) / TPB, TPB, 0, stream>>>(ei, ew, dinv, nrm);

    // ---- weights -> bf16 transposed ----
    k_convW<<<(HID * FIN + TPB - 1) / TPB, TPB, 0, stream>>>(W1, W1t, FIN, HID, HID);
    k_convW<<<(NPAD * HID + TPB - 1) / TPB, TPB, 0, stream>>>(W2, W2t, HID, NCLS, NPAD);

    // ---- layer 1 ----
    dim3 g1((NN + 127) / 128, HID / 64);
    k_gemm_bf16<<<g1, 256, 0, stream>>>(x, W1t, t1, NN, FIN, HID);
    k_selfloop_init<<<((size_t)NN * HID + TPB - 1) / TPB, TPB, 0, stream>>>(t1, dinv, h1, HID);
    k_scatter<<<EE, HID, 0, stream>>>(t1, h1, ei, nrm, HID, HID);
    k_bias_relu<<<((size_t)NN * HID + TPB - 1) / TPB, TPB, 0, stream>>>(h1, b1);

    // ---- layer 2 ----
    dim3 g2((NN + 127) / 128, NPAD / 64);
    k_gemm_bf16<<<g2, 256, 0, stream>>>(h1, W2t, t2, NN, HID, NPAD);
    k_selfloop_init<<<((size_t)NN * NPAD + TPB - 1) / TPB, TPB, 0, stream>>>(t2, dinv, ag2, NPAD);
    k_scatter<<<EE, NPAD, 0, stream>>>(t2, ag2, ei, nrm, NPAD, NCLS);
    k_logsoftmax<<<(NN + TPB - 1) / TPB, TPB, 0, stream>>>(ag2, b2, out);
}
